// EfficientAdaptivePiecewiseConv2d_22351009808400
// MI455X (gfx1250) — compile-verified, matrix operand order checked by reading, not run on hardware
//
#include <hip/hip_runtime.h>
#include <hip/hip_bf16.h>

typedef __attribute__((ext_vector_type(16))) __bf16 v16bf;
typedef __attribute__((ext_vector_type(8)))  float  v8f;

#define HH   128
#define WW   128
#define CIN  64
#define CEXP 192
#define COUT 128
#define NBAT 16

// workspace layout (bf16 elements):
//   expanded NHWC: [16][128][128][192]        -> 100,663,296 B  (fits in 192MB L2)
//   wt:            [9][128][192] (tap, n, k)  ->     442,368 B
static constexpr size_t EXP_ELEMS = (size_t)NBAT * HH * WW * CEXP;

// dynamic LDS layout (u16 elements), double buffered:
//   A: [2][130][40]   (130-px halo strip x 32ch, stride 40 = conflict-free)
//   B: [2][3][128][40]
#define A_ELEMS (2 * 130 * 40)
#define B_ELEMS (2 * 3 * 128 * 40)
#define SMEM_BYTES ((A_ELEMS + B_ELEMS) * 2)

// low 32 bits of a generic pointer into LDS == wave-relative LDS byte address
__device__ __forceinline__ unsigned lds_addr32(const void* p) {
    return (unsigned)(unsigned long long)p;
}

__device__ __forceinline__ void async_load_b128(unsigned lds_off, unsigned gbyte_off,
                                                const void* sbase) {
    asm volatile("global_load_async_to_lds_b128 %0, %1, %2"
                 :: "v"(lds_off), "v"(gbyte_off), "s"(sbase) : "memory");
}

// ---------------------------------------------------------------------------
// Kernel 1: channel mean -> 3-knot piecewise basis -> bf16 "expanded" (NHWC)
//   x values register-cached: single pass over HBM (67 MB instead of 134 MB)
// ---------------------------------------------------------------------------
__global__ __launch_bounds__(256)
void expand_kernel(const float* __restrict__ x, __hip_bfloat16* __restrict__ eo) {
    int p   = blockIdx.x * blockDim.x + threadIdx.x;   // pixel id, 0..262143
    int b   = p >> 14;
    int rem = p & 16383;
    int y   = rem >> 7;
    int xx  = rem & 127;
    const float* xb = x + ((size_t)b * CIN) * (HH * WW) + y * WW + xx;

    float xv[CIN];
    float s = 0.f;
#pragma unroll
    for (int c = 0; c < CIN; ++c) {
        xv[c] = xb[c * (HH * WW)];
        s += xv[c];
    }
    float m = s * (1.0f / 64.0f);

    // knots {-1, 0, 1}, replicating reference masking exactly
    float w0 = (m <= 0.f) ? -m : 0.f;
    float lh = (m >= -1.f && m <= 0.f) ? (m + 1.f) : 0.f;
    float rh = (m >   0.f && m <= 1.f) ? (1.f - m) : 0.f;
    float w1 = lh + rh;
    float w2 = (m >= 0.f) ? m : 0.f;

    __hip_bfloat16* o = eo + (size_t)p * CEXP;
#pragma unroll
    for (int cb = 0; cb < CIN; cb += 8) {
        union { __hip_bfloat16 h[24]; uint4 q[3]; } u;
#pragma unroll
        for (int j = 0; j < 8; ++j) {
            float v = xv[cb + j];
            u.h[j * 3 + 0] = __float2bfloat16(v * w0);
            u.h[j * 3 + 1] = __float2bfloat16(v * w1);
            u.h[j * 3 + 2] = __float2bfloat16(v * w2);
        }
        uint4* o4 = (uint4*)(o + cb * 3);
        o4[0] = u.q[0]; o4[1] = u.q[1]; o4[2] = u.q[2];
    }
}

// ---------------------------------------------------------------------------
// Kernel 2: repack W (OIHW fp32) -> wt[tap][n][k] bf16, k contiguous
// ---------------------------------------------------------------------------
__global__ __launch_bounds__(256)
void wpack_kernel(const float* __restrict__ W, __hip_bfloat16* __restrict__ wt) {
    int i = blockIdx.x * blockDim.x + threadIdx.x;     // flat OIHW index
    if (i >= COUT * CEXP * 9) return;
    int kw = i % 3;  int t = i / 3;
    int kh = t % 3;  t /= 3;
    int kk = t % CEXP;
    int n  = t / CEXP;
    wt[(((size_t)(kh * 3 + kw) * COUT + n) * CEXP) + kk] = __float2bfloat16(W[i]);
}

// ---------------------------------------------------------------------------
// Kernel 3: implicit-GEMM 3x3 conv, v_wmma_f32_16x16x32_bf16
//   block = 256 thr (8 waves), tile M=128 px (full row) x N=128 Cout
//   wave (wm 0..3, wn 0..1): 32 px x 64 cout -> 2x4 accumulators
//   software pipeline over (kh,c0): async-stage batch j+1 (exactly 8
//   global_load_async_to_lds_b128 per wave) while computing batch j;
//   drain with s_wait_asynccnt 8.  Edge kh taps are skipped (contribution 0);
//   halo rows (x=-1,128) are statically zeroed once.
// ---------------------------------------------------------------------------
struct Frag32B { uint4 lo, hi; };

__global__ __launch_bounds__(256)
void conv_wmma_kernel(const __hip_bfloat16* __restrict__ expanded,
                      const __hip_bfloat16* __restrict__ wt,
                      float* __restrict__ out) {
    extern __shared__ __align__(16) unsigned short smem[];
    unsigned short* sA = smem;             // [2][130][40]
    unsigned short* sB = smem + A_ELEMS;   // [2][3][128][40]

    const int tid = threadIdx.x;
    const int y   = blockIdx.x & 127;
    const int b   = blockIdx.x >> 7;

    const int wave = tid >> 5;
    const int wm   = wave >> 1;       // 0..3 : 32-pixel slab
    const int wn   = wave & 1;        // 0..1 : 64-cout slab
    const int lane = tid & 31;
    const int half = lane >> 4;
    const int row  = lane & 15;

    const v8f vzero = {0.f, 0.f, 0.f, 0.f, 0.f, 0.f, 0.f, 0.f};
    v8f acc[2][4];
#pragma unroll
    for (int mi = 0; mi < 2; ++mi)
#pragma unroll
        for (int ni = 0; ni < 4; ++ni) acc[mi][ni] = vzero;

    // valid kh range (contiguous); edge taps contribute exactly zero
    const int khs = (y == 0)      ? 1 : 0;
    const int khe = (y == HH - 1) ? 1 : 2;
    const int nit = (khe - khs + 1) * 6;

    // statically zero halo rows 0 and 129 of both A buffers (5 uint4 per row)
    if (tid < 20) {
        int r   = tid / 5;                       // 0..3
        int bi  = r >> 1;
        int hr  = (r & 1) ? 129 : 0;
        int off = (tid % 5) * 8;
        *(uint4*)&sA[(bi * 130 + hr) * 40 + off] = make_uint4(0u, 0u, 0u, 0u);
    }

    // stage batch j into buffer bi: 2 A-loads + 6 B-loads per thread (uniform)
    auto stage = [&](int j, int bi) {
        int kh = khs + j / 6;
        int c0 = (j % 6) * 32;
        int yy = y + kh - 1;                      // guaranteed in range
        unsigned arow = (unsigned)((((size_t)b * HH + yy) * WW) * CEXP * 2);
#pragma unroll
        for (int i = 0; i < 2; ++i) {
            int idx  = tid + i * 256;             // 0..511
            int s    = 1 + (idx >> 2);            // strip row 1..128 (x = s-1)
            int koff = (idx & 3) * 8;
            unsigned lo = lds_addr32(&sA[(bi * 130 + s) * 40 + koff]);
            unsigned go = arow + (unsigned)(((s - 1) * CEXP + c0 + koff) * 2);
            async_load_b128(lo, go, (const void*)expanded);
        }
#pragma unroll
        for (int i = 0; i < 6; ++i) {
            int idx  = tid + i * 256;             // 0..1535
            int t    = idx >> 9;                  // 0..2 (kw)
            int n    = (idx >> 2) & 127;
            int koff = (idx & 3) * 8;
            unsigned lo = lds_addr32(&sB[((bi * 3 + t) * 128 + n) * 40 + koff]);
            unsigned go = (unsigned)(((((kh * 3 + t) * COUT + n) * CEXP)
                                      + c0 + koff) * 2);
            async_load_b128(lo, go, (const void*)wt);
        }
    };

    // 24 WMMA per wave on buffer bi
    auto domath = [&](int bi) {
#pragma unroll
        for (int kw = 0; kw < 3; ++kw) {
            v16bf aF[2];
#pragma unroll
            for (int mi = 0; mi < 2; ++mi) {
                int r = wm * 32 + mi * 16 + row + kw;     // 0..129
                const unsigned short* p = &sA[(bi * 130 + r) * 40];
                Frag32B f;
                f.lo = *(const uint4*)(p + half * 8);
                f.hi = *(const uint4*)(p + 16 + half * 8);
                aF[mi] = __builtin_bit_cast(v16bf, f);
            }
#pragma unroll
            for (int ni = 0; ni < 4; ++ni) {
                int r = wn * 64 + ni * 16 + row;
                const unsigned short* p = &sB[((bi * 3 + kw) * 128 + r) * 40];
                Frag32B f;
                f.lo = *(const uint4*)(p + half * 8);
                f.hi = *(const uint4*)(p + 16 + half * 8);
                v16bf bF = __builtin_bit_cast(v16bf, f);
#pragma unroll
                for (int mi = 0; mi < 2; ++mi)
                    acc[mi][ni] = __builtin_amdgcn_wmma_f32_16x16x32_bf16(
                        false, aF[mi], false, bF,
                        (short)0, acc[mi][ni], false, false);
            }
        }
    };

    // software pipeline
    stage(0, 0);
    for (int j = 0; j < nit; ++j) {
        int bi = j & 1;
        if (j + 1 < nit) {
            stage(j + 1, bi ^ 1);
            asm volatile("s_wait_asynccnt 8" ::: "memory");   // batch j done
        } else {
            asm volatile("s_wait_asynccnt 0" ::: "memory");
        }
        __syncthreads();          // publish batch j across waves
        domath(bi);
        __syncthreads();          // all reads of buf bi done before reuse
    }

    // ---- store: lane holds 8 consecutive x positions for fixed cout ----
#pragma unroll
    for (int mi = 0; mi < 2; ++mi)
#pragma unroll
        for (int ni = 0; ni < 4; ++ni) {
            int cout = wn * 64 + ni * 16 + row;
            int px   = wm * 32 + mi * 16 + half * 8;
            float* dst = out + ((((size_t)b * COUT + cout) * HH + y) * WW + px);
            v8f c = acc[mi][ni];
            float4 v0, v1;
            v0.x = c[0]; v0.y = c[1]; v0.z = c[2]; v0.w = c[3];
            v1.x = c[4]; v1.y = c[5]; v1.z = c[6]; v1.w = c[7];
            *(float4*)dst       = v0;
            *(float4*)(dst + 4) = v1;
        }
}

// ---------------------------------------------------------------------------
extern "C" void kernel_launch(void* const* d_in, const int* in_sizes, int n_in,
                              void* d_out, int out_size, void* d_ws, size_t ws_size,
                              hipStream_t stream) {
    const float* x = (const float*)d_in[0];
    const float* W = (const float*)d_in[1];
    float* out = (float*)d_out;

    __hip_bfloat16* exp_buf = (__hip_bfloat16*)d_ws;
    __hip_bfloat16* wt      = exp_buf + EXP_ELEMS;

    const int npix = NBAT * HH * WW;                    // 262144
    expand_kernel<<<npix / 256, 256, 0, stream>>>(x, exp_buf);

    const int nw = COUT * CEXP * 9;                     // 221184
    wpack_kernel<<<(nw + 255) / 256, 256, 0, stream>>>(W, wt);

    // 16 batches * 128 rows = 2048 blocks; full row x full Cout per block
    conv_wmma_kernel<<<NBAT * HH, 256, SMEM_BYTES, stream>>>(exp_buf, wt, out);
}
